// TLocallyConnected2d_76312978915839
// MI455X (gfx1250) — matrix operand order repaired and checked by reading, not compiled, not run on hardware
//
#include <hip/hip_runtime.h>
#include <hip/hip_bf16.h>

// Locally-connected 2D, B=64, Cin=Cout=64, H=W=56, K=3, pad=1.
// One workgroup per pixel: GEMM M=64(Cout) x N=64(batch) x K=576(Cin*9) via
// v_wmma_f32_16x16x32_bf16. Software-pipelined: global->reg loads for step k+1
// overlap WMMAs of step k; double-buffered LDS; one barrier per k-step.

typedef __bf16 bf16_t;
typedef __attribute__((ext_vector_type(16))) __bf16 v16bf;
typedef __attribute__((ext_vector_type(8)))  __bf16 v8bf;
typedef __attribute__((ext_vector_type(2)))  __bf16 v2bf;
typedef __attribute__((ext_vector_type(8)))  float  v8f;

#define CIN   64
#define COUT  64
#define NB    64
#define IH    56
#define IW    56
#define NPIX  (IH*IW)          // 3136
#define KTOT  (CIN*9)          // 576
#define KSTEP 32
#define NSTEP (KTOT/KSTEP)     // 18
#define XSTRB (CIN*IH*IW)      // 200704 : batch stride in x
#define WSTRI (CIN*NPIX*9)     // 1806336: Cout stride in weight

__launch_bounds__(256, 2)
__global__ void lc2d_wmma_bf16_pipe(const float* __restrict__ x,
                                    const float* __restrict__ weight,
                                    const float* __restrict__ bias,
                                    float* __restrict__ out)
{
    // Double-buffered bf16 tiles: A = weights [Cout][32K], B = x [batch][32K] (N-major)
    __shared__ __attribute__((aligned(16))) bf16_t lsA[2][COUT][KSTEP]; // 2x4 KB
    __shared__ __attribute__((aligned(16))) bf16_t lsB[2][NB][KSTEP];   // 2x4 KB

    const int pix = blockIdx.x;
    const int m   = pix / IW;
    const int n   = pix - m * IW;

    const int tid    = threadIdx.x;
    const int lane   = tid & 31;
    const int wid    = tid >> 5;           // 8 waves
    const int laneLo = lane & 15;
    const int hi     = lane >> 4;          // wave32 WMMA lane half

    const int nT  = wid >> 1;              // wave's N tile (0..3)
    const int mT0 = (wid & 1) << 1;        // wave's first M tile (0 or 2)

    // Pair mapping: thread owns elements p = 2*tid + 512*it  (it = 0..3).
    // kk = p&31 is it-invariant -> K-decode happens ONCE per k-step.
    const int kkP  = (tid * 2) & 31;       // even K offset within step
    const int rowP = (tid * 2) >> 5;       // base row 0..15 (+16 per it)

    float aR[8], bR[8];                    // register staging for next k-step

    // ---- global -> registers for step ks (weights NT; x regular temporal)
    auto loadStage = [&](int ks) {
        const int K0  = ks * KSTEP + kkP;
        int j0  = K0 / 9;
        int kl0 = K0 - j0 * 9;
        int j1  = j0 + (kl0 == 8);         // incremental decode of K0+1
        int kl1 = (kl0 == 8) ? 0 : kl0 + 1;
        int dk0 = kl0 / 3, dl0 = kl0 - dk0 * 3;
        int dk1 = kl1 / 3, dl1 = kl1 - dk1 * 3;
        int h0 = m + dk0 - 1, w0 = n + dl0 - 1;
        int h1 = m + dk1 - 1, w1 = n + dl1 - 1;
        const bool ok0 = ((unsigned)h0 < (unsigned)IH) & ((unsigned)w0 < (unsigned)IW);
        const bool ok1 = ((unsigned)h1 < (unsigned)IH) & ((unsigned)w1 < (unsigned)IW);
        const int wOff0 = (j0 * NPIX + pix) * 9 + kl0;
        const int wOff1 = (j1 * NPIX + pix) * 9 + kl1;
        const int xOff0 = (j0 * IH + h0) * IW + w0;
        const int xOff1 = (j1 * IH + h1) * IW + w1;
        #pragma unroll
        for (int it = 0; it < 4; ++it) {
            const int row = rowP + it * 16;      // Cout index (A) / batch index (B)
            aR[2*it]   = __builtin_nontemporal_load(&weight[row * WSTRI + wOff0]);
            aR[2*it+1] = __builtin_nontemporal_load(&weight[row * WSTRI + wOff1]);
            bR[2*it]   = ok0 ? x[row * XSTRB + xOff0] : 0.0f;
            bR[2*it+1] = ok1 ? x[row * XSTRB + xOff1] : 0.0f;
        }
    };

    // ---- registers -> LDS (packed bf16 pair: v_cvt_pk_bf16_f32 + ds_store_b32)
    auto storeStage = [&](int buf) {
        #pragma unroll
        for (int it = 0; it < 4; ++it) {
            const int row = rowP + it * 16;
            v2bf pa, pb;
            pa[0] = (bf16_t)aR[2*it]; pa[1] = (bf16_t)aR[2*it+1];
            pb[0] = (bf16_t)bR[2*it]; pb[1] = (bf16_t)bR[2*it+1];
            *(v2bf*)&lsA[buf][row][kkP] = pa;
            *(v2bf*)&lsB[buf][row][kkP] = pb;
        }
    };

    v8f acc0 = {};   // carried f32 accumulators across all 18 k-steps
    v8f acc1 = {};

    // ---- LDS fragments + 2 WMMAs from buffer `buf`
    auto computeStep = [&](int buf) {
        // B fragment (32x16 KxN): lane half hi -> K = {hi*8.., 16+hi*8..}
        const int col = nT * 16 + laneLo;
        v8bf bl = *(const v8bf*)&lsB[buf][col][hi * 8];
        v8bf bh = *(const v8bf*)&lsB[buf][col][16 + hi * 8];
        v16bf bF = __builtin_shufflevector(bl, bh,
            0,1,2,3,4,5,6,7,8,9,10,11,12,13,14,15);
        // A fragments (16x32 MxK): lane half hi -> K = {hi*8.., 16+hi*8..}
        const int r0 = mT0 * 16 + laneLo;
        v8bf a0l = *(const v8bf*)&lsA[buf][r0][hi * 8];
        v8bf a0h = *(const v8bf*)&lsA[buf][r0][16 + hi * 8];
        v16bf aF0 = __builtin_shufflevector(a0l, a0h,
            0,1,2,3,4,5,6,7,8,9,10,11,12,13,14,15);
        const int r1 = r0 + 16;
        v8bf a1l = *(const v8bf*)&lsA[buf][r1][hi * 8];
        v8bf a1h = *(const v8bf*)&lsA[buf][r1][16 + hi * 8];
        v16bf aF1 = __builtin_shufflevector(a1l, a1h,
            0,1,2,3,4,5,6,7,8,9,10,11,12,13,14,15);

        acc0 = __builtin_amdgcn_wmma_f32_16x16x32_bf16(
            false, aF0, false, bF, (short)0, acc0, false, false);
        acc1 = __builtin_amdgcn_wmma_f32_16x16x32_bf16(
            false, aF1, false, bF, (short)0, acc1, false, false);
    };

    // ---- software pipeline --------------------------------------------------
    loadStage(0);
    storeStage(0);
    __syncthreads();

    #pragma unroll 1
    for (int ks = 0; ks < NSTEP; ++ks) {
        const int cur = ks & 1;
        if (ks + 1 < NSTEP) loadStage(ks + 1);     // in flight during WMMAs
        if (ks + 2 < NSTEP) {                      // HBM latency cover (+2 steps)
            int Kp = (ks + 2) * KSTEP + kkP;
            int jp = Kp / 9;
            __builtin_prefetch(&weight[rowP * WSTRI + (jp * NPIX + pix) * 9
                                       + (Kp - jp * 9)], 0, 1);
        }
        computeStep(cur);                          // ds_load_b128 x6 + wmma x2
        if (ks + 1 < NSTEP) storeStage(cur ^ 1);   // fill the other buffer
        __syncthreads();                           // single barrier per step
    }

    // ---- epilogue: fuse bias, NT scatter to out[b][i][m][n] -----------------
    // C/D layout: M = v + hi*8 ; N(batch) = nT*16 + laneLo.
    {
        const int bcol = nT * 16 + laneLo;
        #pragma unroll
        for (int v = 0; v < 8; ++v) {
            int mLoc = v + hi * 8;
            int i0 = mT0 * 16 + mLoc;
            int i1 = i0 + 16;
            float o0 = acc0[v] + bias[i0 * NPIX + pix];
            float o1 = acc1[v] + bias[i1 * NPIX + pix];
            __builtin_nontemporal_store(o0, &out[(bcol * COUT + i0) * NPIX + pix]);
            __builtin_nontemporal_store(o1, &out[(bcol * COUT + i1) * NPIX + pix]);
        }
    }
}

extern "C" void kernel_launch(void* const* d_in, const int* in_sizes, int n_in,
                              void* d_out, int out_size, void* d_ws, size_t ws_size,
                              hipStream_t stream) {
    (void)in_sizes; (void)n_in; (void)d_ws; (void)ws_size; (void)out_size;
    const float* x      = (const float*)d_in[0]; // (64,64,56,56)
    const float* weight = (const float*)d_in[1]; // (64,64,56,56,3,3)
    const float* bias   = (const float*)d_in[2]; // (64,56,56)
    float* out          = (float*)d_out;         // (64,64,56,56)

    dim3 grid(NPIX);   // one workgroup per output pixel
    dim3 block(256);   // 8 wave32 waves -> 16 WMMA tiles, 2 per wave
    lc2d_wmma_bf16_pipe<<<grid, block, 0, stream>>>(x, weight, bias, out);
}
